// MemoryBank_28286654611819
// MI455X (gfx1250) — matrix operand order repaired and checked by reading, not compiled, not run on hardware
//
#include <hip/hip_runtime.h>
#include <hip/hip_bf16.h>
#include <math.h>

typedef __bf16 bf16;
typedef __attribute__((ext_vector_type(4)))  __bf16 v4bf;
typedef __attribute__((ext_vector_type(8)))  __bf16 v8bf;
typedef __attribute__((ext_vector_type(16))) __bf16 v16bf;
typedef __attribute__((ext_vector_type(8)))  float  v8f;

#define D_DIM   1024
#define B_ROWS  4096
#define M_ROWS  32768
#define TOPK    8
#define CHUNK   1024
#define NCHUNKS (M_ROWS / CHUNK)   // 32

static __device__ __forceinline__ v16bf cat8(v8bf lo, v8bf hi) {
  return __builtin_shufflevector(lo, hi, 0,1,2,3,4,5,6,7,8,9,10,11,12,13,14,15);
}

// A fragment (16x32 bf16, ISA layout): lo lanes hold K 0..7 & 16..23, hi lanes K 8..15 & 24..31
static __device__ __forceinline__ v16bf load_a_global(const bf16* __restrict__ A, int lda,
                                                      int rowBase, int k0, int m, int half) {
  const bf16* p = A + (size_t)(rowBase + m) * lda + k0 + half * 8;
  v8bf lo = *(const v8bf*)p;
  v8bf hi = *(const v8bf*)(p + 16);
  return cat8(lo, hi);
}

// B fragment (32x16 bf16): column n = row of W (C = A @ W^T); lo lanes K 0..7 & 16..23, hi lanes K 8..15 & 24..31
static __device__ __forceinline__ v16bf load_b_global(const bf16* __restrict__ W, int ldw,
                                                      int colBase, int k0, int n, int half) {
  const bf16* p = W + (size_t)(colBase + n) * ldw + k0 + half * 8;
  v8bf lo = *(const v8bf*)p;
  v8bf hi = *(const v8bf*)(p + 16);
  return cat8(lo, hi);
}

// ---------------- f32 -> bf16 conversion ----------------
__global__ __launch_bounds__(256)
void f32_to_bf16_kernel(const float* __restrict__ in, bf16* __restrict__ out, int n) {
  int i = (blockIdx.x * 256 + threadIdx.x) * 4;
  if (i + 3 < n) {
    float4 v = *(const float4*)(in + i);
    v4bf o = { (bf16)v.x, (bf16)v.y, (bf16)v.z, (bf16)v.w };
    *(v4bf*)(out + i) = o;
  }
}

// ---------------- Projection GEMM: C[M x N] = A[M x D] @ W^T[N x D] + bias ----------------
// 8 waves/WG, each wave computes one 16x16 C tile; WG covers 32 rows x 64 cols.
// W tile staged through LDS (shared by the 4 waves of a row-strip).
template<bool OUT_BF16>
__global__ __launch_bounds__(256)
void proj_gemm_kernel(const bf16* __restrict__ A, const bf16* __restrict__ W,
                      const float* __restrict__ bias,
                      float* __restrict__ outF, bf16* __restrict__ outB,
                      int M, int N) {
  __shared__ bf16 ldsW[64 * 32];     // 4 KB: 64 cols x 32 K
  const int tid  = threadIdx.x;
  const int wave = tid >> 5;
  const int lane = tid & 31;
  const int half = lane >> 4;
  const int l16  = lane & 15;
  const int rowBase = blockIdx.x * 32 + (wave >> 2) * 16;
  const int colBlk  = blockIdx.y * 64;
  const int colLoc  = (wave & 3) * 16;
  const int col     = colBlk + colLoc + l16;

  v8f acc = {};
  for (int k0 = 0; k0 < D_DIM; k0 += 32) {
    // stage W[colBlk..+64)[k0..+32) into LDS: 256 threads x 8 bf16 each
    {
      int n  = tid >> 2;            // 0..63
      int kk = (tid & 3) * 8;       // 0,8,16,24
      *(v8bf*)&ldsW[n * 32 + kk] = *(const v8bf*)&W[(size_t)(colBlk + n) * D_DIM + k0 + kk];
      if (k0 + 32 < D_DIM)
        __builtin_prefetch(&W[(size_t)(colBlk + n) * D_DIM + k0 + 32 + kk], 0, 3);
    }
    __syncthreads();
    v16bf a = load_a_global(A, D_DIM, rowBase, k0, l16, half);
    const bf16* bp = &ldsW[(colLoc + l16) * 32 + half * 8];
    v16bf b = cat8(*(const v8bf*)bp, *(const v8bf*)(bp + 16));
    acc = __builtin_amdgcn_wmma_f32_16x16x32_bf16(false, a, false, b, (short)0, acc, false, false);
    __syncthreads();
  }

  float bv = bias ? bias[col] : 0.0f;
  #pragma unroll
  for (int i = 0; i < 8; ++i) {
    int row = rowBase + i + half * 8;
    float v = acc[i] + bv;
    if (OUT_BF16) outB[(size_t)row * N + col] = (bf16)v;
    else          outF[(size_t)row * N + col] = v;
  }
}

// ---------------- Fused sims GEMM + per-chunk top-8 ----------------
// WG owns 16 query rows x 1024 mem columns. Each wave register-blocks a
// 16x128 strip with 8 accumulators: per K-step, ONE A fragment feeds 8 WMMAs
// (8x A reuse, 8 independent WMMAs to hide load latency). Scaled sims are
// deposited into a 64 KB LDS tile, then 16 lanes run per-row top-8 insertion.
__global__ __launch_bounds__(256)
void sims_topk_kernel(const bf16* __restrict__ Q,   // [B x D] bf16
                      const bf16* __restrict__ K,   // [M x D] bf16
                      float* __restrict__ candV, int* __restrict__ candI) {
  extern __shared__ float simsLds[];   // 16 * CHUNK floats = 64 KB
  const int tid  = threadIdx.x;
  const int wave = tid >> 5;
  const int lane = tid & 31;
  const int half = lane >> 4;
  const int l16  = lane & 15;
  const int rowBase   = blockIdx.x * 16;
  const int chunkBase = blockIdx.y * CHUNK;
  const int colLoc    = wave * 128;          // wave's 128-column strip
  const float scale   = 1.0f / 32.0f;        // 1/sqrt(1024), TEMP = 1

  v8f acc[8] = {};
  for (int k0 = 0; k0 < D_DIM; k0 += 32) {
    v16bf a = load_a_global(Q, D_DIM, rowBase, k0, l16, half);   // reused by 8 tiles
    #pragma unroll
    for (int t = 0; t < 8; ++t) {
      v16bf b = load_b_global(K, D_DIM, chunkBase + colLoc + t * 16, k0, l16, half);
      acc[t] = __builtin_amdgcn_wmma_f32_16x16x32_bf16(false, a, false, b,
                                                       (short)0, acc[t], false, false);
    }
  }

  #pragma unroll
  for (int t = 0; t < 8; ++t) {
    #pragma unroll
    for (int i = 0; i < 8; ++i) {
      int r = i + half * 8;
      simsLds[r * CHUNK + colLoc + t * 16 + l16] = acc[t][i] * scale;
    }
  }
  __syncthreads();

  if (tid < 16) {
    const int r = tid;
    float tv[TOPK]; int ti[TOPK];
    #pragma unroll
    for (int j = 0; j < TOPK; ++j) { tv[j] = -1e30f; ti[j] = 0; }
    for (int c = 0; c < CHUNK; ++c) {
      float v = simsLds[r * CHUNK + c];
      if (v > tv[TOPK - 1]) {
        int pos = TOPK - 1;
        while (pos > 0 && v > tv[pos - 1]) {
          tv[pos] = tv[pos - 1]; ti[pos] = ti[pos - 1]; --pos;
        }
        tv[pos] = v; ti[pos] = chunkBase + c;
      }
    }
    size_t base = ((size_t)(rowBase + r) * NCHUNKS + blockIdx.y) * TOPK;
    #pragma unroll
    for (int j = 0; j < TOPK; ++j) { candV[base + j] = tv[j]; candI[base + j] = ti[j]; }
  }
}

// ---------------- Candidate reduction: 256 candidates/row -> final top-8 indices ----------------
__global__ __launch_bounds__(256)
void topk_reduce_kernel(const float* __restrict__ candV, const int* __restrict__ candI,
                        int* __restrict__ topi) {
  int b = blockIdx.x * 256 + threadIdx.x;
  if (b >= B_ROWS) return;
  float tv[TOPK]; int ti[TOPK];
  #pragma unroll
  for (int j = 0; j < TOPK; ++j) { tv[j] = -1e30f; ti[j] = 0; }
  size_t base = (size_t)b * NCHUNKS * TOPK;
  for (int c = 0; c < NCHUNKS * TOPK; ++c) {
    float v = candV[base + c];
    if (v > tv[TOPK - 1]) {
      int iv = candI[base + c];
      int pos = TOPK - 1;
      while (pos > 0 && v > tv[pos - 1]) {
        tv[pos] = tv[pos - 1]; ti[pos] = ti[pos - 1]; --pos;
      }
      tv[pos] = v; ti[pos] = iv;
    }
  }
  #pragma unroll
  for (int j = 0; j < TOPK; ++j) topi[b * TOPK + j] = ti[j];
}

// ---------------- Gather V rows + LayerNorm ----------------
__global__ __launch_bounds__(256)
void gather_ln_kernel(const float* __restrict__ V, const int* __restrict__ topi,
                      const float* __restrict__ gamma, const float* __restrict__ beta,
                      float* __restrict__ out) {
  __shared__ float redS[8];
  __shared__ float redQ[8];
  const int row = blockIdx.x;            // b*8 + j
  const int idx = topi[row];
  const int tid = threadIdx.x;
  const float4 v = ((const float4*)(V + (size_t)idx * D_DIM))[tid];

  float s  = v.x + v.y + v.z + v.w;
  float ss = v.x * v.x + v.y * v.y + v.z * v.z + v.w * v.w;
  #pragma unroll
  for (int off = 16; off > 0; off >>= 1) {
    s  += __shfl_down(s,  off, 32);
    ss += __shfl_down(ss, off, 32);
  }
  const int wave = tid >> 5, lane = tid & 31;
  if (lane == 0) { redS[wave] = s; redQ[wave] = ss; }
  __syncthreads();
  if (tid == 0) {
    float S = 0.f, SS = 0.f;
    #pragma unroll
    for (int w = 0; w < 8; ++w) { S += redS[w]; SS += redQ[w]; }
    redS[0] = S; redQ[0] = SS;
  }
  __syncthreads();
  const float mu  = redS[0] * (1.0f / D_DIM);
  const float var = redQ[0] * (1.0f / D_DIM) - mu * mu;
  const float inv = rsqrtf(var + 1e-5f);

  const float4 g  = ((const float4*)gamma)[tid];
  const float4 bt = ((const float4*)beta)[tid];
  float4 o;
  o.x = (v.x - mu) * inv * g.x + bt.x;
  o.y = (v.y - mu) * inv * g.y + bt.y;
  o.z = (v.z - mu) * inv * g.z + bt.z;
  o.w = (v.w - mu) * inv * g.w + bt.w;
  ((float4*)(out + (size_t)row * D_DIM))[tid] = o;
}

// ---------------- host launcher ----------------
extern "C" void kernel_launch(void* const* d_in, const int* in_sizes, int n_in,
                              void* d_out, int out_size, void* d_ws, size_t ws_size,
                              hipStream_t stream) {
  (void)in_sizes; (void)n_in; (void)out_size; (void)ws_size;
  const float* query = (const float*)d_in[0];
  const float* mem   = (const float*)d_in[1];
  const float* Wq    = (const float*)d_in[2];
  const float* bq    = (const float*)d_in[3];
  const float* Wk    = (const float*)d_in[4];
  const float* bk    = (const float*)d_in[5];
  const float* Wv    = (const float*)d_in[6];
  const float* bv    = (const float*)d_in[7];
  const float* gamma = (const float*)d_in[8];
  const float* beta  = (const float*)d_in[9];
  // d_in[10] is k == 8 (compile-time TOPK)

  char* ws = (char*)d_ws;
  size_t off = 0;
  auto alloc = [&](size_t bytes) -> void* {
    void* p = ws + off;
    off += (bytes + 255) & ~(size_t)255;
    return p;
  };

  bf16*  qb   = (bf16*)alloc((size_t)B_ROWS * D_DIM * sizeof(bf16));
  bf16*  memb = (bf16*)alloc((size_t)M_ROWS * D_DIM * sizeof(bf16));
  bf16*  wqb  = (bf16*)alloc((size_t)D_DIM * D_DIM * sizeof(bf16));
  bf16*  wkb  = (bf16*)alloc((size_t)D_DIM * D_DIM * sizeof(bf16));
  bf16*  wvb  = (bf16*)alloc((size_t)D_DIM * D_DIM * sizeof(bf16));
  bf16*  qpb  = (bf16*)alloc((size_t)B_ROWS * D_DIM * sizeof(bf16));  // projected Q (bf16)
  bf16*  kpb  = (bf16*)alloc((size_t)M_ROWS * D_DIM * sizeof(bf16));  // projected K (bf16)
  float* vp   = (float*)alloc((size_t)M_ROWS * D_DIM * sizeof(float)); // projected V (f32)
  float* candV = (float*)alloc((size_t)B_ROWS * NCHUNKS * TOPK * sizeof(float));
  int*   candI = (int*)  alloc((size_t)B_ROWS * NCHUNKS * TOPK * sizeof(int));
  int*   topi  = (int*)  alloc((size_t)B_ROWS * TOPK * sizeof(int));

  // 1) f32 -> bf16 conversions
  f32_to_bf16_kernel<<<(B_ROWS * D_DIM) / 1024, 256, 0, stream>>>(query, qb,   B_ROWS * D_DIM);
  f32_to_bf16_kernel<<<(M_ROWS * D_DIM) / 1024, 256, 0, stream>>>(mem,   memb, M_ROWS * D_DIM);
  f32_to_bf16_kernel<<<(D_DIM * D_DIM) / 1024, 256, 0, stream>>>(Wq, wqb, D_DIM * D_DIM);
  f32_to_bf16_kernel<<<(D_DIM * D_DIM) / 1024, 256, 0, stream>>>(Wk, wkb, D_DIM * D_DIM);
  f32_to_bf16_kernel<<<(D_DIM * D_DIM) / 1024, 256, 0, stream>>>(Wv, wvb, D_DIM * D_DIM);

  // 2) projections via bf16 WMMA
  dim3 gq(B_ROWS / 32, D_DIM / 64);
  dim3 gm(M_ROWS / 32, D_DIM / 64);
  proj_gemm_kernel<true ><<<gq, 256, 0, stream>>>(qb,   wqb, bq, nullptr, qpb, B_ROWS, D_DIM);
  proj_gemm_kernel<true ><<<gm, 256, 0, stream>>>(memb, wkb, bk, nullptr, kpb, M_ROWS, D_DIM);
  proj_gemm_kernel<false><<<gm, 256, 0, stream>>>(memb, wvb, bv, vp, nullptr, M_ROWS, D_DIM);

  // 3) fused sims GEMM + per-chunk top-8 (64 KB LDS sims tile)
  dim3 gs(B_ROWS / 16, NCHUNKS);
  sims_topk_kernel<<<gs, 256, 16 * CHUNK * sizeof(float), stream>>>(qpb, kpb, candV, candI);

  // 4) reduce candidates to final top-8 indices
  topk_reduce_kernel<<<B_ROWS / 256, 256, 0, stream>>>(candV, candI, topi);

  // 5) gather V rows + LayerNorm into d_out [B, 8, D]
  gather_ln_kernel<<<B_ROWS * TOPK, 256, 0, stream>>>(vp, topi, gamma, beta, (float*)d_out);
}
